// ChannelClustering_17935783428605
// MI455X (gfx1250) — matrix operand order repaired and checked by reading, not compiled, not run on hardware
//
#include <hip/hip_runtime.h>
#include <hip/hip_bf16.h>

typedef __attribute__((ext_vector_type(16))) _Float16 v16h;
typedef __attribute__((ext_vector_type(8)))  float    v8f;

#define NB   16      // batch
#define NC   512     // channels
#define NS   4096    // spatial (64*64)
#define NK   8       // clusters
#define KM_ITERS 20
#define KSTEP 32     // K per WMMA stage (f16 16x16x32)

// ---------------------------------------------------------------------------
// K1: fused per-(b,c) mean over spatial dim + f32 -> f16 image copy.
// The f16 copy halves GEMM-side memory traffic and lets the GEMM stage tiles
// with async LDS DMA (no conversion needed in the hot loop).
// grid (NC, NB), block 256
// ---------------------------------------------------------------------------
__global__ __launch_bounds__(256) void cc_mean_cvt_kernel(const float* __restrict__ img,
                                                          float* __restrict__ meanv,
                                                          _Float16* __restrict__ f16img) {
    const int c = blockIdx.x, b = blockIdx.y;
    const size_t rowoff = ((size_t)b * NC + c) * NS;
    const float* p = img + rowoff;
    _Float16*    q = f16img + rowoff;
    float s = 0.f;
    for (int e = threadIdx.x; e < NS; e += 256) {
        const float v = p[e];
        s += v;
        q[e] = (_Float16)v;     // RTE, matches WMMA rounding rules
    }
    __shared__ float red[256];
    red[threadIdx.x] = s;
    __syncthreads();
    for (int w = 128; w > 0; w >>= 1) {
        if ((int)threadIdx.x < w) red[threadIdx.x] += red[threadIdx.x + w];
        __syncthreads();
    }
    if (threadIdx.x == 0) meanv[b * NC + c] = red[0] / (float)NS;
}

// ---------------------------------------------------------------------------
// K2: raw Gram G[b] = F[b] * F[b]^T (F = [512,4096], f16 copy) via f16 WMMA.
// 128x128 tile per block, 8 waves (wave32), each wave 32x64 = 2x4 WMMA tiles.
// Tiles staged with CDNA5 async LDS DMA (global_load_async_to_lds_b128,
// ASYNCcnt), double-buffered so DMA of tile i+1 overlaps WMMAs of tile i.
// grid (4, 4, NB), block 256
// ---------------------------------------------------------------------------
__global__ __launch_bounds__(256) void cc_gram_wmma_kernel(const _Float16* __restrict__ F16,
                                                           float* __restrict__ G) {
    const int bn = blockIdx.x, bm = blockIdx.y, b = blockIdx.z;

    // stride 40 halves = 80 B: rows 16B-aligned for b128 async stores
    __shared__ _Float16 ldsA[2][128][40];
    __shared__ _Float16 ldsB[2][128][40];

    const int tid  = threadIdx.x;
    const int lane = tid & 31;
    const int wave = tid >> 5;
    const int wr   = wave & 3;    // wave row -> M offset wr*32
    const int wc   = wave >> 2;   // wave col -> N offset wc*64
    const int ln   = lane & 15;
    const int lg   = lane >> 4;   // lane group (0/1)

    v8f acc[2][4];
#pragma unroll
    for (int mt = 0; mt < 2; ++mt)
#pragma unroll
        for (int nt = 0; nt < 4; ++nt)
#pragma unroll
            for (int r = 0; r < 8; ++r) acc[mt][nt][r] = 0.f;

    const size_t base = (size_t)b * NC * NS;
    const int lrow = tid >> 1;                   // 2 threads per tile row
    const int kk   = (tid & 1) * 16;             // 16 contiguous halves (32 B) per thread
    const _Float16* gA = F16 + base + (size_t)(bm * 128 + lrow) * NS + kk;
    const _Float16* gB = F16 + base + (size_t)(bn * 128 + lrow) * NS + kk;

    constexpr int NSTEPS = NS / KSTEP;           // 128

    // per-lane LDS byte offsets (generic LDS pointer truncates to LDS offset)
    unsigned laOff[2], lbOff[2];
#pragma unroll
    for (int bi = 0; bi < 2; ++bi) {
        laOff[bi] = (unsigned)(uintptr_t)&ldsA[bi][lrow][kk];
        lbOff[bi] = (unsigned)(uintptr_t)&ldsB[bi][lrow][kk];
    }

    // async-stage one 128x32-half tile pair into buffer bi for K-slice k0.
    // INST_OFFSET is added to BOTH the LDS and global addresses (ISA 10.7).
    auto issue_tiles = [&](int bi, int k0) {
        const unsigned long long ga = (unsigned long long)(uintptr_t)(gA + k0);
        const unsigned long long gb = (unsigned long long)(uintptr_t)(gB + k0);
        asm volatile("global_load_async_to_lds_b128 %0, %1, off"
                     :: "v"(laOff[bi]), "v"(ga) : "memory");
        asm volatile("global_load_async_to_lds_b128 %0, %1, off offset:16"
                     :: "v"(laOff[bi]), "v"(ga) : "memory");
        asm volatile("global_load_async_to_lds_b128 %0, %1, off"
                     :: "v"(lbOff[bi]), "v"(gb) : "memory");
        asm volatile("global_load_async_to_lds_b128 %0, %1, off offset:16"
                     :: "v"(lbOff[bi]), "v"(gb) : "memory");
    };

    issue_tiles(0, 0);   // prologue

    for (int step = 0; step < NSTEPS; ++step) {
        const int cur = step & 1;
        // own async loads for buffer `cur` complete, then block-wide visibility
        asm volatile("s_wait_asynccnt 0x0" ::: "memory");
        __syncthreads();
        if (step + 1 < NSTEPS) issue_tiles(cur ^ 1, (step + 1) * KSTEP);

        // ---- load fragments per CDNA5 wave32 WMMA layouts ----
        v16h af[2], bf[4];
#pragma unroll
        for (int mt = 0; mt < 2; ++mt) {
            const int m = wr * 32 + mt * 16 + ln;   // A: lane 0-15 -> M; lanes 16-31 same M
#pragma unroll
            for (int r = 0; r < 8; ++r) {
#pragma unroll
                for (int p = 0; p < 2; ++p) {
                    const int K = (r < 4 ? (2 * r + p) : (16 + 2 * (r - 4) + p)) + 8 * lg;
                    af[mt][2 * r + p] = ldsA[cur][m][K];
                }
            }
        }
#pragma unroll
        for (int nt = 0; nt < 4; ++nt) {
            const int n = wc * 64 + nt * 16 + ln;   // B: lane -> N; K = 8*lanegroup + (h < 8 ? h : h + 8)
            const _Float16* pb = &ldsB[cur][n][lg * 8];
#pragma unroll
            for (int h = 0; h < 16; ++h) bf[nt][h] = pb[h < 8 ? h : h + 8];
        }

        // ---- 8 WMMAs per wave per K-step ----
#pragma unroll
        for (int mt = 0; mt < 2; ++mt)
#pragma unroll
            for (int nt = 0; nt < 4; ++nt)
                acc[mt][nt] = __builtin_amdgcn_wmma_f32_16x16x32_f16(
                    false, af[mt], false, bf[nt], (short)0, acc[mt][nt], false, false);
    }

    // ---- epilogue: C/D layout -> G ----
    float* Gb = G + (size_t)b * NC * NC;
#pragma unroll
    for (int mt = 0; mt < 2; ++mt) {
        const int m0 = bm * 128 + wr * 32 + mt * 16;
#pragma unroll
        for (int nt = 0; nt < 4; ++nt) {
            const int n0 = bn * 128 + wc * 64 + nt * 16;
#pragma unroll
            for (int r = 0; r < 8; ++r) {
                const int m = m0 + r + 8 * lg;     // VGPR r -> M=r (lanes 0-15), M=r+8 (16-31)
                Gb[(size_t)m * NC + n0 + ln] = acc[mt][nt][r];
            }
        }
    }
}

// ---------------------------------------------------------------------------
// K3: diagonal pass -> invmag (centered norm^-1) and sq (raw |x|^2)
// ---------------------------------------------------------------------------
__global__ __launch_bounds__(256) void cc_diag_kernel(const float* __restrict__ G,
                                                      const float* __restrict__ meanv,
                                                      float* __restrict__ invmag,
                                                      float* __restrict__ sq) {
    const int i = blockIdx.x * 256 + threadIdx.x;
    if (i >= NB * NC) return;
    const int b = i / NC, c = i % NC;
    const float g  = G[(size_t)b * NC * NC + (size_t)c * NC + c];
    const float m  = meanv[i];
    const float Cii = g - (float)NS * m * m;
    invmag[i] = rsqrtf(fmaxf(Cii, 1e-30f));
    sq[i]     = g;
}

// ---------------------------------------------------------------------------
// K4: corr[b,i,j] = (G_ij - S*m_i*m_j) * invmag_i * invmag_j
// ---------------------------------------------------------------------------
__global__ __launch_bounds__(256) void cc_corr_kernel(const float* __restrict__ G,
                                                      const float* __restrict__ meanv,
                                                      const float* __restrict__ invmag,
                                                      float* __restrict__ corr) {
    const size_t id = (size_t)blockIdx.x * 256 + threadIdx.x;
    if (id >= (size_t)NB * NC * NC) return;
    const int j = (int)(id % NC);
    const int i = (int)((id / NC) % NC);
    const int b = (int)(id / ((size_t)NC * NC));
    const float g = G[id];
    corr[id] = (g - (float)NS * meanv[b * NC + i] * meanv[b * NC + j])
               * invmag[b * NC + i] * invmag[b * NC + j];
}

// ---------------------------------------------------------------------------
// K5: Lloyd k-means per batch (k=8, 20 iters, init = first 8 rows)
// grid NB blocks, block 256 (8 waves). Centers + accumulators in LDS.
// ---------------------------------------------------------------------------
__global__ __launch_bounds__(256) void cc_kmeans_kernel(const float* __restrict__ corr,
                                                        int* __restrict__ labels) {
    const int b   = blockIdx.x;
    const int tid = threadIdx.x;
    __shared__ float cent[NK][NC];
    __shared__ float accm[NK][NC];
    __shared__ float cnt[NK];

    const float* cb = corr + (size_t)b * NC * NC;

    for (int e = tid; e < NK * NC; e += 256) cent[e / NC][e % NC] = cb[e];
    __syncthreads();

    for (int it = 0; it < KM_ITERS; ++it) {
        for (int e = tid; e < NK * NC; e += 256) accm[e / NC][e % NC] = 0.f;
        if (tid < NK) cnt[tid] = 0.f;
        __syncthreads();

        for (int ci = tid; ci < NC; ci += 256) {
            const float* row = cb + (size_t)ci * NC;
            float d[NK];
#pragma unroll
            for (int k = 0; k < NK; ++k) d[k] = 0.f;
            for (int j = 0; j < NC; ++j) {
                const float x = row[j];
#pragma unroll
                for (int k = 0; k < NK; ++k) {
                    const float t = x - cent[k][j];
                    d[k] += t * t;
                }
            }
            int best = 0;
            float bv = d[0];
#pragma unroll
            for (int k = 1; k < NK; ++k) { if (d[k] < bv) { bv = d[k]; best = k; } }
            atomicAdd(&cnt[best], 1.f);
            for (int j = 0; j < NC; ++j) atomicAdd(&accm[best][j], row[j]);
        }
        __syncthreads();

        for (int e = tid; e < NK * NC; e += 256) {
            const int k = e / NC, j = e % NC;
            const float c = cnt[k];
            if (c > 0.f) cent[k][j] = accm[k][j] / c;   // keep old center if empty
        }
        __syncthreads();
    }

    for (int ci = tid; ci < NC; ci += 256) {
        const float* row = cb + (size_t)ci * NC;
        float d[NK];
#pragma unroll
        for (int k = 0; k < NK; ++k) d[k] = 0.f;
        for (int j = 0; j < NC; ++j) {
            const float x = row[j];
#pragma unroll
            for (int k = 0; k < NK; ++k) {
                const float t = x - cent[k][j];
                d[k] += t * t;
            }
        }
        int best = 0;
        float bv = d[0];
#pragma unroll
        for (int k = 1; k < NK; ++k) { if (d[k] < bv) { bv = d[k]; best = k; } }
        labels[b * NC + ci] = best;
    }
}

// ---------------------------------------------------------------------------
// K6: medoid selection. D_ij computed on the fly from G and sq.
// Per-cluster argmin with lowest-index tie-break via packed u64 ds_min.
// grid NB blocks, block 512
// ---------------------------------------------------------------------------
__global__ __launch_bounds__(512) void cc_medoid_kernel(const float* __restrict__ G,
                                                        const float* __restrict__ sq,
                                                        const int* __restrict__ labels,
                                                        int* __restrict__ idxbuf) {
    const int b = blockIdx.x;
    const int i = threadIdx.x;   // one thread per channel
    __shared__ float sqs[NC];
    __shared__ int   lab[NC];
    __shared__ float cntS[NK];
    __shared__ unsigned long long best[NK];

    if (i < NK) { cntS[i] = 0.f; best[i] = ~0ull; }
    __syncthreads();

    sqs[i] = sq[b * NC + i];
    const int myl = labels[b * NC + i];
    lab[i] = myl;
    atomicAdd(&cntS[myl], 1.f);
    __syncthreads();

    const float* Gr = G + (size_t)b * NC * NC + (size_t)i * NC;
    const float sqi = sqs[i];
    float a[NK];
#pragma unroll
    for (int k = 0; k < NK; ++k) a[k] = 0.f;
    for (int j = 0; j < NC; ++j) {
        const float d = sqrtf(fmaxf(sqi + sqs[j] - 2.f * Gr[j], 0.f));
        a[lab[j]] += d;
    }
    const float avg = a[myl] / fmaxf(cntS[myl], 1.f);
    const unsigned long long key =
        ((unsigned long long)__float_as_uint(avg) << 32) | (unsigned)i;  // avg>=0: bits monotone
    atomicMin(&best[myl], key);
    __syncthreads();

    if (i < NK) {
        unsigned idx = (unsigned)(best[i] & 0xffffffffu);
        if (idx >= NC) idx = 0;          // empty cluster: argmin over all-inf -> 0
        idxbuf[b * NK + i] = (int)idx;
    }
}

// ---------------------------------------------------------------------------
// K7: gather medoid channels -> output [NB, NK, 64, 64] (from f32 image)
// ---------------------------------------------------------------------------
__global__ __launch_bounds__(256) void cc_gather_kernel(const float* __restrict__ img,
                                                        const int* __restrict__ idxbuf,
                                                        float* __restrict__ out) {
    const int blk = blockIdx.x;           // b*NK + k
    const int b   = blk / NK;
    const int idx = idxbuf[blk];
    const float* src = img + ((size_t)b * NC + idx) * NS;
    float* dst = out + (size_t)blk * NS;
    for (int e = threadIdx.x; e < NS; e += 256) dst[e] = src[e];
}

// ---------------------------------------------------------------------------
extern "C" void kernel_launch(void* const* d_in, const int* in_sizes, int n_in,
                              void* d_out, int out_size, void* d_ws, size_t ws_size,
                              hipStream_t stream) {
    const float* img = (const float*)d_in[0];
    float* out = (float*)d_out;

    // workspace layout (bytes); total ~101 MB
    char* ws = (char*)d_ws;
    size_t off = 0;
    _Float16* F16 = (_Float16*)(ws + off); off += (size_t)NB * NC * NS * sizeof(_Float16); // 67 MB
    float* G      = (float*)(ws + off); off += (size_t)NB * NC * NC * sizeof(float);       // 16.8 MB
    float* corr   = (float*)(ws + off); off += (size_t)NB * NC * NC * sizeof(float);       // 16.8 MB
    float* meanv  = (float*)(ws + off); off += (size_t)NB * NC * sizeof(float);
    float* invmag = (float*)(ws + off); off += (size_t)NB * NC * sizeof(float);
    float* sq     = (float*)(ws + off); off += (size_t)NB * NC * sizeof(float);
    int*   labels = (int*)  (ws + off); off += (size_t)NB * NC * sizeof(int);
    int*   idxbuf = (int*)  (ws + off); off += (size_t)NB * NK * sizeof(int);

    cc_mean_cvt_kernel<<<dim3(NC, NB), 256, 0, stream>>>(img, meanv, F16);
    cc_gram_wmma_kernel<<<dim3(NC / 128, NC / 128, NB), 256, 0, stream>>>(F16, G);
    cc_diag_kernel<<<(NB * NC + 255) / 256, 256, 0, stream>>>(G, meanv, invmag, sq);
    {
        const size_t n = (size_t)NB * NC * NC;
        cc_corr_kernel<<<(unsigned)((n + 255) / 256), 256, 0, stream>>>(G, meanv, invmag, corr);
    }
    cc_kmeans_kernel<<<NB, 256, 0, stream>>>(corr, labels);
    cc_medoid_kernel<<<NB, 512, 0, stream>>>(G, sq, labels, idxbuf);
    cc_gather_kernel<<<NB * NK, 256, 0, stream>>>(img, idxbuf, out);
    (void)in_sizes; (void)n_in; (void)out_size; (void)ws_size;
}
